// DragonCurveAttention_30511447670982
// MI455X (gfx1250) — compile-verified, matrix operand order checked by reading, not run on hardware
//
#include <hip/hip_runtime.h>

// ---------------- problem constants ----------------
constexpr int BATCH  = 2;
constexpr int SEQ    = 2048;
constexpr int DMODEL = 1024;
constexpr int HEADS  = 16;
constexpr int HD     = 64;     // head dim (qk and v)

typedef __attribute__((ext_vector_type(16))) __bf16 v16bf;
typedef __attribute__((ext_vector_type(8)))  float  v8f;
typedef __attribute__((ext_vector_type(4)))  int    v4i_async;

union FragU { v16bf v; unsigned short u[16]; };

#if defined(__gfx1250__) && __has_builtin(__builtin_amdgcn_global_load_async_to_lds_b128)
#define ASYNC_LDS 1
#else
#define ASYNC_LDS 0
#endif

__device__ __forceinline__ void async_wait0() {
#if __has_builtin(__builtin_amdgcn_s_wait_asynccnt)
  __builtin_amdgcn_s_wait_asynccnt(0);
#else
  asm volatile("s_wait_asynccnt 0x0" ::: "memory");
#endif
}

// fp32 -> bf16 via the hardware convert
__device__ __forceinline__ unsigned short f2bf(float f) {
  union { __bf16 b; unsigned short u; } c;
  c.b = (__bf16)f;
  return c.u;
}
// pack two fp32 -> one dword of two bf16 (v_cvt_pk_bf16_f32)
__device__ __forceinline__ unsigned int pack2bf(float lo, float hi) {
  union { __bf16 b[2]; unsigned int u; } c;
  c.b[0] = (__bf16)lo;
  c.b[1] = (__bf16)hi;
  return c.u;
}

__device__ __forceinline__ v8f wmma_bf16(v16bf a, v16bf b, v8f c) {
  // (neg_a, A, neg_b, B, c_mod, C, reuse_a, reuse_b)
  return __builtin_amdgcn_wmma_f32_16x16x32_bf16(false, a, false, b, (short)0, c,
                                                 false, false);
}

// Fragment load (16x32 tile) from row-major source p[m*ld + kofs + k].
// ISA layout: lanes 0-15 hold row m=lane with K {0..7,16..23}; lanes 16-31 same
// row with K {8..15,24..31}.  Contiguous 8-element runs -> 2x ds_load_b128.
// Used for A fragments and for B fragments whose source is stored N-major
// (rows = N, cols = K), i.e. everything in this file.
__device__ __forceinline__ v16bf load_frag_row(const unsigned short* p, int ld, int kofs) {
  const int lane = threadIdx.x & 31;
  const int half = lane >> 4;
  const int m    = lane & 15;
  FragU f;
#pragma unroll
  for (int e = 0; e < 16; ++e) {
    const int k = ((e >> 3) << 4) + (half << 3) + (e & 7);
    f.u[e] = p[m * ld + kofs + k];
  }
  return f.v;
}

__device__ __forceinline__ float rowmax16(float x) {
#pragma unroll
  for (int m = 1; m < 16; m <<= 1) x = fmaxf(x, __shfl_xor(x, m, 32));
  return x;
}
__device__ __forceinline__ float rowsum16(float x) {
#pragma unroll
  for (int m = 1; m < 16; m <<= 1) x += __shfl_xor(x, m, 32);
  return x;
}

// ================= kernel 1: QKV projection =================
// C = (x @ W) * scale, bf16 output.
// dmajor==0: head-major [B,H,S,64] (Q, K).  dmajor==1: dim-major [B,H,64,S] (V),
// so the attention kernel can stage V tiles with pure vector copies.
// Block: 256 thr (8 waves). Tile: 128(M) x 64(N); blockIdx.x == head.
__global__ __launch_bounds__(256)
void qkv_proj_kernel(const float* __restrict__ x, const float* __restrict__ W,
                     unsigned short* __restrict__ out, float scale, int dmajor) {
  __shared__ unsigned short sA[128 * 32];   // x tile, row-major (M x K)
  __shared__ unsigned short sBt[64 * 32];   // W tile, N-major   (N x K)
  const int tid  = threadIdx.x;
  const int wave = tid >> 5;
  const int lane = tid & 31;
  const int n0   = blockIdx.x * 64;
  const int m0   = blockIdx.y * 128;

  v8f acc[4] = {};

  for (int kc = 0; kc < DMODEL; kc += 32) {
    __syncthreads();
    {  // x tile: 128 x 32 fp32 -> bf16 pairs stored straight to LDS
      const int r = tid >> 1, c0 = (tid & 1) * 16;
      const float* src = x + (size_t)(m0 + r) * DMODEL + kc + c0;
      unsigned int* dstw = (unsigned int*)(sA + r * 32 + c0);
#pragma unroll
      for (int i = 0; i < 8; ++i)
        dstw[i] = pack2bf(src[2 * i], src[2 * i + 1]);
      if (kc + 32 < DMODEL) __builtin_prefetch(src + 32, 0, 1);
    }
    {  // W tile: 32(K) x 64(N) fp32, stored transposed (N-major)
      const int r = tid >> 3, c0 = (tid & 7) * 8;   // r = k row, c0 = n col
      const float* src = W + (size_t)(kc + r) * DMODEL + n0 + c0;
#pragma unroll
      for (int i = 0; i < 8; ++i) sBt[(c0 + i) * 32 + r] = f2bf(src[i]);
      if (kc + 32 < DMODEL) __builtin_prefetch(src + 32 * DMODEL, 0, 1);
    }
    __syncthreads();

    const v16bf a = load_frag_row(sA + wave * 16 * 32, 32, 0);
#pragma unroll
    for (int nt = 0; nt < 4; ++nt) {
      const v16bf b = load_frag_row(sBt + nt * 16 * 32, 32, 0);
      acc[nt] = wmma_bf16(a, b, acc[nt]);
    }
  }

  const int h  = blockIdx.x;
  const int n  = lane & 15;
  const int mb = (lane >> 4) * 8;
#pragma unroll
  for (int nt = 0; nt < 4; ++nt)
#pragma unroll
    for (int v = 0; v < 8; ++v) {
      const int m  = m0 + wave * 16 + mb + v;
      const int bb = m / SEQ, s = m % SEQ;
      const int d  = nt * 16 + n;
      const unsigned short val = f2bf(acc[nt][v] * scale);
      if (dmajor)
        out[(((size_t)bb * HEADS + h) * HD + d) * SEQ + s] = val;     // [B,H,64,S]
      else
        out[(((size_t)bb * HEADS + h) * SEQ + s) * HD + d] = val;     // [B,H,S,64]
    }
}

// -------- K/V chunk staging (32 keys) for attention --------
// K source: [S,64] row-major -> tile is one contiguous 4KB block.
// V source: [64,S] dim-major -> 64 rows of 32 contiguous keys, staged N-major.
__device__ __forceinline__ void stage_kv(const unsigned short* __restrict__ Kh,
                                         const unsigned short* __restrict__ Vh,
                                         unsigned short* sK, unsigned short* sVt,
                                         int kc, int tid) {
  const int vd = tid >> 2, vc0 = (tid & 3) * 8;
#if ASYNC_LDS
  __builtin_amdgcn_global_load_async_to_lds_b128(
      (__attribute__((address_space(1))) v4i_async*)(Kh + (size_t)kc * HD + tid * 8),
      (__attribute__((address_space(3))) v4i_async*)(sK + tid * 8), 0, 0);
  __builtin_amdgcn_global_load_async_to_lds_b128(
      (__attribute__((address_space(1))) v4i_async*)(Vh + (size_t)vd * SEQ + kc + vc0),
      (__attribute__((address_space(3))) v4i_async*)(sVt + vd * 32 + vc0), 0, 0);
#else
  ((uint4*)sK)[tid] = ((const uint4*)(Kh + (size_t)kc * HD))[tid];
  *(uint4*)(sVt + vd * 32 + vc0) = *(const uint4*)(Vh + (size_t)vd * SEQ + kc + vc0);
#endif
}

// ================= kernel 2: flash attention =================
// One block per (b,h, 128-query tile); 8 waves, 16 query rows each.
// Double-buffered K/V chunks (async global->LDS), online softmax, fractal
// per-key weight applied to scores pre-softmax.
__global__ __launch_bounds__(256)
void attn_kernel(const unsigned short* __restrict__ Q,
                 const unsigned short* __restrict__ Kb,
                 const unsigned short* __restrict__ Vb,
                 const float* __restrict__ fw,
                 unsigned short* __restrict__ Ob) {
  __shared__ unsigned short sK[2][32 * HD];    // 2 x 4 KB, row-major (key x d)
  __shared__ unsigned short sV[2][HD * 32];    // 2 x 4 KB, N-major   (d x key)
  __shared__ unsigned short sP[8 * 16 * 32];   // 8 KB, per-wave P scratch

  const int tid  = threadIdx.x;
  const int wave = tid >> 5;
  const int lane = tid & 31;
  const int bh   = blockIdx.y;                 // b*H + h
  const int q0   = blockIdx.x * 128 + wave * 16;

  const unsigned short* Qh = Q  + (size_t)bh * SEQ * HD;
  const unsigned short* Kh = Kb + (size_t)bh * SEQ * HD;
  const unsigned short* Vh = Vb + (size_t)bh * HD * SEQ;   // dim-major

  // Q fragments (16 rows x 64) stay in registers; 1/sqrt(d) already folded in.
  const v16bf qa0 = load_frag_row(Qh + (size_t)q0 * HD, HD, 0);
  const v16bf qa1 = load_frag_row(Qh + (size_t)q0 * HD, HD, 32);

  v8f acc[4] = {};
  float mstate[8], lstate[8];
#pragma unroll
  for (int v = 0; v < 8; ++v) { mstate[v] = -1e30f; lstate[v] = 0.0f; }

  const int n  = lane & 15;
  const int mb = (lane >> 4) * 8;
  unsigned short* myP = sP + wave * 16 * 32;

  constexpr int NCHUNK = SEQ / 32;
  stage_kv(Kh, Vh, sK[0], sV[0], 0, tid);
#if ASYNC_LDS
  async_wait0();
#endif
  __syncthreads();

  for (int j = 0; j < NCHUNK; ++j) {
    const int kc = j * 32;
    const unsigned short* cK = sK[j & 1];
    const unsigned short* cV = sV[j & 1];

    // prefetch next chunk into the other buffer while we compute
    if (j + 1 < NCHUNK)
      stage_kv(Kh, Vh, sK[(j + 1) & 1], sV[(j + 1) & 1], kc + 32, tid);

    // scores: S[16q x 32k] = Q(16x64) @ K^T(64x32)
    v8f s0 = {}, s1 = {};
    s0 = wmma_bf16(qa0, load_frag_row(cK,           HD, 0),  s0);
    s0 = wmma_bf16(qa1, load_frag_row(cK,           HD, 32), s0);
    s1 = wmma_bf16(qa0, load_frag_row(cK + 16 * HD, HD, 0),  s1);
    s1 = wmma_bf16(qa1, load_frag_row(cK + 16 * HD, HD, 32), s1);

    const float w0 = fw[kc + n];
    const float w1 = fw[kc + 16 + n];

#pragma unroll
    for (int v = 0; v < 8; ++v) {
      const float a0 = s0[v] * w0;
      const float a1 = s1[v] * w1;
      const float rmax = rowmax16(fmaxf(a0, a1));
      const float nm   = fmaxf(mstate[v], rmax);
      const float corr = __expf(mstate[v] - nm);
      const float p0   = __expf(a0 - nm);
      const float p1   = __expf(a1 - nm);
      const float rs   = rowsum16(p0 + p1);
      lstate[v] = lstate[v] * corr + rs;
      mstate[v] = nm;
      acc[0][v] *= corr; acc[1][v] *= corr; acc[2][v] *= corr; acc[3][v] *= corr;
      myP[(mb + v) * 32 + n]      = f2bf(p0);
      myP[(mb + v) * 32 + 16 + n] = f2bf(p1);
    }

    // O += P(16x32) @ V(32x64); V staged N-major so B frags are contiguous
    const v16bf pa = load_frag_row(myP, 32, 0);
#pragma unroll
    for (int nt = 0; nt < 4; ++nt) {
      const v16bf vb = load_frag_row(cV + nt * 16 * 32, 32, 0);
      acc[nt] = wmma_bf16(pa, vb, acc[nt]);
    }

#if ASYNC_LDS
    async_wait0();
#endif
    __syncthreads();
  }

  // normalize, store bf16 attn output in [B,S,H*64] layout for out-proj
  const int bb = bh / HEADS;
  const int h  = bh % HEADS;
  float inv[8];
#pragma unroll
  for (int v = 0; v < 8; ++v) inv[v] = 1.0f / lstate[v];
#pragma unroll
  for (int nt = 0; nt < 4; ++nt)
#pragma unroll
    for (int v = 0; v < 8; ++v) {
      const int sq = q0 + mb + v;
      Ob[((size_t)bb * SEQ + sq) * DMODEL + h * HD + nt * 16 + n] =
          f2bf(acc[nt][v] * inv[v]);
    }
}

// ================= kernel 3: output projection =================
// out(fp32) = attn(bf16 [4096,1024]) @ Wo(fp32->bf16)
__global__ __launch_bounds__(256)
void out_proj_kernel(const unsigned short* __restrict__ A,
                     const float* __restrict__ W, float* __restrict__ out) {
  __shared__ unsigned short sA[128 * 32];
  __shared__ unsigned short sBt[64 * 32];   // Wo tile, N-major
  const int tid  = threadIdx.x;
  const int wave = tid >> 5;
  const int lane = tid & 31;
  const int n0   = blockIdx.x * 64;
  const int m0   = blockIdx.y * 128;

  v8f acc[4] = {};

  for (int kc = 0; kc < DMODEL; kc += 32) {
    __syncthreads();
    {  // A tile already bf16: vector copy 128x32
      const int r = tid >> 1, c0 = (tid & 1) * 16;
      const uint4* src = (const uint4*)(A + (size_t)(m0 + r) * DMODEL + kc + c0);
      uint4* dst = (uint4*)(sA + r * 32 + c0);
      dst[0] = src[0];
      dst[1] = src[1];
      if (kc + 32 < DMODEL) __builtin_prefetch((const char*)src + 64, 0, 1);
    }
    {  // Wo tile: 32 x 64 fp32 -> bf16, stored N-major
      const int r = tid >> 3, c0 = (tid & 7) * 8;
      const float* src = W + (size_t)(kc + r) * DMODEL + n0 + c0;
#pragma unroll
      for (int i = 0; i < 8; ++i) sBt[(c0 + i) * 32 + r] = f2bf(src[i]);
      if (kc + 32 < DMODEL) __builtin_prefetch(src + 32 * DMODEL, 0, 1);
    }
    __syncthreads();

    const v16bf a = load_frag_row(sA + wave * 16 * 32, 32, 0);
#pragma unroll
    for (int nt = 0; nt < 4; ++nt) {
      const v16bf b = load_frag_row(sBt + nt * 16 * 32, 32, 0);
      acc[nt] = wmma_bf16(a, b, acc[nt]);
    }
  }

  const int nn = lane & 15;
  const int mb = (lane >> 4) * 8;
#pragma unroll
  for (int nt = 0; nt < 4; ++nt)
#pragma unroll
    for (int v = 0; v < 8; ++v) {
      const int m = m0 + wave * 16 + mb + v;
      out[(size_t)m * DMODEL + n0 + nt * 16 + nn] = acc[nt][v];
    }
}

// ================= host launcher =================
extern "C" void kernel_launch(void* const* d_in, const int* in_sizes, int n_in,
                              void* d_out, int out_size, void* d_ws, size_t ws_size,
                              hipStream_t stream) {
  (void)in_sizes; (void)n_in; (void)out_size; (void)ws_size;
  const float* x  = (const float*)d_in[0];
  const float* Wq = (const float*)d_in[1];
  const float* Wk = (const float*)d_in[2];
  const float* Wv = (const float*)d_in[3];
  const float* Wo = (const float*)d_in[4];
  const float* fw = (const float*)d_in[5];
  float* out = (float*)d_out;

  const size_t headElems = (size_t)BATCH * HEADS * SEQ * HD;  // 4M elems
  unsigned short* qws = (unsigned short*)d_ws;                // 8 MB
  unsigned short* kws = qws + headElems;                      // 8 MB
  unsigned short* vws = kws + headElems;                      // 8 MB
  unsigned short* aws = vws + headElems;                      // 8 MB

  const dim3 blk(256);
  const dim3 gProj(HEADS, (BATCH * SEQ) / 128);               // 16 x 32
  qkv_proj_kernel<<<gProj, blk, 0, stream>>>(x, Wq, qws, 0.125f, 0);  // 1/sqrt(64)
  qkv_proj_kernel<<<gProj, blk, 0, stream>>>(x, Wk, kws, 1.0f, 0);
  qkv_proj_kernel<<<gProj, blk, 0, stream>>>(x, Wv, vws, 1.0f, 1);    // dim-major

  const dim3 gAttn(SEQ / 128, BATCH * HEADS);                 // 16 x 32
  attn_kernel<<<gAttn, blk, 0, stream>>>(qws, kws, vws, fw, aws);

  const dim3 gOut(DMODEL / 64, (BATCH * SEQ) / 128);          // 16 x 32
  out_proj_kernel<<<gOut, blk, 0, stream>>>(aws, Wo, out);
}
